// RESNET_SNN_STDB_50586124812453
// MI455X (gfx1250) — matrix operand order repaired and checked by reading, not compiled, not run on hardware
//
#include <hip/hip_runtime.h>

// ---------------------------------------------------------------------------
// Spiking ResNet forward for MI455X (gfx1250), compile-only target.
//
// Roofline: ~4.5 GMAC/timestep x 8 steps; bf16 weights (~10MB) are L2-resident
// (192MB L2); conv arithmetic intensity 576..4608 MAC/output -> matrix-core
// bound -> v_wmma_f32_16x16x32_bf16, fp32 accumulate. Spikes are exactly
// {0,1,1.25} -> bf16 activations are lossless; LIF membrane state is fp32.
//
// v4.1: async builtin exists (per round-4 diagnostic) and takes int*-typed
// pointers -> cast global/LDS addresses to AS1/AS3 int*.
// ---------------------------------------------------------------------------

typedef __attribute__((ext_vector_type(16))) __bf16 v16bf;
typedef __attribute__((ext_vector_type(8)))  __bf16 v8bf;
typedef __attribute__((ext_vector_type(8)))  float  v8f;

#if __has_builtin(__builtin_amdgcn_global_load_async_to_lds_b32)
#define HAVE_ASYNC_LDS 1
typedef __attribute__((address_space(1))) int __as1i;
typedef __attribute__((address_space(3))) int __as3i;
#else
#define HAVE_ASYNC_LDS 0
#endif

#define BATCH 32
#define TSTEPS 8

__global__ void zero_kernel(float* __restrict__ p, int n) {
  int i = blockIdx.x * blockDim.x + threadIdx.x;
  if (i < n) p[i] = 0.0f;
}

// fp32 -> bf16 with K-padding: dst[co*Kpad + k] = k<K ? src[co*K + k] : 0
__global__ void cvt_w_kernel(const float* __restrict__ src, __bf16* __restrict__ dst,
                             int K, int Kpad, int n) {
  int i = blockIdx.x * blockDim.x + threadIdx.x;
  if (i >= n) return;
  int co = i / Kpad, k = i - co * Kpad;
  dst[i] = (k < K) ? (__bf16)src[co * K + k] : (__bf16)0.0f;
}

// ---------------------------------------------------------------------------
// Implicit-GEMM convolution via WMMA bf16.
//   GEMM view:  M = Cout,  N = B*Hout*Wout,  K = Cin*KH*KW (weights padded
//   to Kpad = ceil64(K) with zeros).
// Block = 256 threads (8 waves, wave32). Wave grid 2(M) x 4(N).
// Block tile = 32(M) x 64(N); K consumed 64 per phase (2 chained WMMAs).
// A: direct global b128 fragment loads (row-major bf16, ISA 10.9).
// B: im2col gather staged to LDS in fragment order
//      lane = n + 16*((k>>3)&1), pos = (k&7) + 8*(k>>4)  -> 32B contiguous per lane,
//    written as (k,k+1) pairs; adjacent in-bounds pairs go via
//    global_load_async_to_lds_b32 when available.
// Spatial sizes are powers of two: Hin=Win=1<<wish, Hout=Wout=1<<wsh.
// ---------------------------------------------------------------------------
template <int KH>
__global__ __launch_bounds__(256)
void conv_wmma_kernel(const __bf16* __restrict__ in, const __bf16* __restrict__ w,
                      float* __restrict__ out,
                      int Cin, int wish, int Cout, int wsh,
                      int stride, int pad, int accumulate) {
  constexpr int KHW = KH * KH;
  const int K     = Cin * KHW;
  const int Kpad  = (K + 63) & ~63;
  const int Hin_  = 1 << wish;
  const int Win_  = Hin_;
  const int Wout_ = 1 << wsh;
  const int hw_sh = 2 * wsh;                 // log2(Hout*Wout)
  const int n0    = blockIdx.x * 64;         // N-tile base (b,ho,wo flattened)
  const int co0   = blockIdx.y * 32;         // M-tile base (output channel)
  const int tid   = threadIdx.x;
  const int lane  = tid & 31;                // wave32
  const int wave  = tid >> 5;
  const int mrow  = wave >> 2;               // 0..1  (which 16 of the 32 M)
  const int ncol  = wave & 3;                // 0..3  (which 16 of the 64 N)

  __shared__ __align__(32) __bf16 b_lds[2][4][32][16];   // [ku][ncol][lane][pos]

  // A-fragment base: lane holds row m = co0 + mrow*16 + (lane&15); its two
  // contiguous K-runs start at 8*(lane>>4) and 16+8*(lane>>4).
  const __bf16* arow =
      w + (size_t)(co0 + mrow * 16 + (lane & 15)) * Kpad + ((lane >> 4) << 3);

  // Per-thread B-gather coordinates: n = tid&63 is invariant across K chunks,
  // so the (b,ho,wo) -> (hi0,wi0,base) decomposition hoists out of the K loop.
  const int    nlB = tid & 63;
  const int    ngB = n0 + nlB;
  const int    bB  = ngB >> hw_sh;
  const int    r2B = ngB & ((1 << hw_sh) - 1);
  const int    hi0 = (r2B >> wsh) * stride - pad;
  const int    wi0 = (r2B & (Wout_ - 1)) * stride - pad;
  const __bf16* inB = in + (((size_t)bB * Cin) << (2 * wish));

  // Scalar gather with bounds/padding mask (returns raw bf16 bits).
  auto fetchB = [&](int kk) -> unsigned {
    unsigned bits = 0;
    if (kk < K) {
      int ci, kh, kw;
      if (KH == 1) { ci = kk; kh = 0; kw = 0; }
      else { ci = kk / KHW; int r = kk - ci * KHW; kh = r / KH; kw = r - kh * KH; }
      int hi = hi0 + kh, wi = wi0 + kw;
      if (hi >= 0 && hi < Hin_ && wi >= 0 && wi < Win_)
        bits = (unsigned)__builtin_bit_cast(
            unsigned short, inB[((((size_t)ci << wish) + hi) << wish) + wi]);
    }
    return bits;
  };

  v8f acc = {};

  const int kChunks = (K + 63) >> 6;         // 64 K per phase
  for (int kc = 0; kc < kChunks; ++kc) {
    const int kbase = kc << 6;

    // ---- stage B: 64(K) x 64(N) gathered conv window, (k,k+1) pairs ----
#pragma unroll
    for (int t = 0; t < 8; ++t) {
      int idx = tid + t * 256;               // 0..2047 pairs
      int ku  = idx >> 10;                   // 0..1
      int k   = ((idx >> 6) & 15) << 1;      // even k in [0,32)
      int kg  = kbase + (ku << 5) + k;
      __bf16* dst = &b_lds[ku][nlB >> 4][(nlB & 15) + (((k >> 3) & 1) << 4)][(k & 7) + ((k >> 4) << 3)];

      bool done = false;
#if HAVE_ASYNC_LDS
      // Fast path: pair is adjacent & fully in-bounds in global memory and
      // 4B-aligned -> async copy global->LDS (no VGPR round trip).
      if (KH == 3 && kg + 1 < K) {
        int ci = kg / KHW; int r = kg - ci * KHW;
        int kh = r / KH;   int kw = r - kh * KH;
        if (kw < KH - 1) {
          int hi = hi0 + kh, wi = wi0 + kw;
          if (hi >= 0 && hi < Hin_ && wi >= 0 && wi + 1 < Win_) {
            const __bf16* gp = &inB[((((size_t)ci << wish) + hi) << wish) + wi];
            if (((size_t)gp & 3) == 0) {
              __builtin_amdgcn_global_load_async_to_lds_b32(
                  (__as1i*)(const void*)gp, (__as3i*)(void*)dst, 0, 0);
              done = true;
            }
          }
        }
      }
#endif
      if (!done) {
        unsigned pk = (fetchB(kg + 1) << 16) | fetchB(kg);
        *reinterpret_cast<unsigned*>(dst) = pk;
      }
    }
#if HAVE_ASYNC_LDS
    // Drain this wave's async global->LDS copies before the block barrier.
#if __has_builtin(__builtin_amdgcn_s_wait_asynccnt)
    __builtin_amdgcn_s_wait_asynccnt(0);
#else
    asm volatile("s_wait_asynccnt 0" ::: "memory");
#endif
#endif
    __syncthreads();

    // prefetch next weight slab (global_prefetch_b8)
    if (kbase + 64 < Kpad) __builtin_prefetch(arow + kbase + 64, 0, 1);

    // ---- two chained WMMAs per phase; A straight from global memory ----
#pragma unroll
    for (int ku = 0; ku < 2; ++ku) {
      const __bf16* ap = arow + kbase + (ku << 5);
      v8bf alo = *reinterpret_cast<const v8bf*>(ap);        // K run 1
      v8bf ahi = *reinterpret_cast<const v8bf*>(ap + 16);   // K run 2
      v16bf af = __builtin_shufflevector(alo, ahi,
                   0, 1, 2, 3, 4, 5, 6, 7, 8, 9, 10, 11, 12, 13, 14, 15);
      v16bf bf = *reinterpret_cast<const v16bf*>(&b_lds[ku][ncol][lane][0]);
      acc = __builtin_amdgcn_wmma_f32_16x16x32_bf16(
          /*neg_a=*/false, af, /*neg_b=*/false, bf,
          /*c_mod=*/(short)0, acc, /*reuse_a=*/false, /*reuse_b=*/false);
    }
    __syncthreads();
  }

  // ---- write D tile: VGPR v -> M = mrow*16 + 8*(lane>>4) + v, N = lane&15.
  // Consecutive v = consecutive output channels -> constant stride Hout*Wout.
  const int ngD   = n0 + ncol * 16 + (lane & 15);
  const int bD    = ngD >> hw_sh;
  const int r2D   = ngD & ((1 << hw_sh) - 1);
  const int mBase = co0 + mrow * 16 + ((lane >> 4) << 3);
  float* po = out + (((size_t)(bD * Cout + mBase) << hw_sh) | r2D);
#pragma unroll
  for (int v = 0; v < 8; ++v) {
    size_t idx = (size_t)v << hw_sh;
    if (accumulate) po[idx] += acc[v];
    else            po[idx]  = acc[v];
  }
}

// ---------------------------------------------------------------------------
// LIF neuron update: mem = leak*mem + (delta [+ res]); spike; soft reset;
// optional dropout mask. Emits bf16 spike map (values {0,1,mask}, lossless).
// ---------------------------------------------------------------------------
__global__ void lif_kernel(float* __restrict__ mem, const float* __restrict__ delta,
                           const __bf16* __restrict__ res, const float* __restrict__ mask,
                           __bf16* __restrict__ out,
                           const float* __restrict__ thr_arr,
                           const float* __restrict__ leak_arr,
                           int layer, int n) {
  int i = blockIdx.x * blockDim.x + threadIdx.x;
  if (i >= n) return;
  float thr  = thr_arr[layer];
  float leak = leak_arr[layer];
  float d = delta[i] + (res ? (float)res[i] : 0.0f);
  float m = leak * mem[i] + d;
  float mth = m / thr - 1.0f;
  float s = (mth > 0.0f) ? 1.0f : 0.0f;
  mem[i] = m - thr * s;
  out[i] = (__bf16)(s * (mask ? mask[i] : 1.0f));
}

__global__ void avgpool2_kernel(const __bf16* __restrict__ in, __bf16* __restrict__ out,
                                int C, int Hi, int Wi, int n) {
  int i = blockIdx.x * blockDim.x + threadIdx.x;
  if (i >= n) return;
  int Ho = Hi >> 1, Wo = Wi >> 1;
  int wo = i % Wo; int t = i / Wo;
  int ho = t % Ho; t /= Ho;
  int c  = t % C;  int b = t / C;
  const __bf16* p = &in[((b * C + c) * Hi + ho * 2) * Wi + wo * 2];
  out[i] = (__bf16)(0.25f * ((float)p[0] + (float)p[1] +
                             (float)p[Wi] + (float)p[Wi + 1]));
}

// Final classifier accumulation: mfc[b,o] += spk[b,:2048] . w_fc[o,:2048]
__global__ void fc_accum_kernel(const __bf16* __restrict__ spk,
                                const float* __restrict__ wfc,
                                float* __restrict__ mfc) {
  int i = blockIdx.x * blockDim.x + threadIdx.x;
  if (i >= BATCH * 10) return;
  int b = i / 10, o = i - b * 10;
  const __bf16* a = &spk[b * 2048];
  const float*  w = &wfc[o * 2048];
  float s = 0.0f;
  for (int k = 0; k < 2048; ++k) s += (float)a[k] * w[k];
  mfc[i] += s;
}

// ---------------------------------------------------------------------------
extern "C" void kernel_launch(void* const* d_in, const int* in_sizes, int n_in,
                              void* d_out, int out_size, void* d_ws, size_t ws_size,
                              hipStream_t stream) {
  const float* x      = (const float*)d_in[0];
  const float* w_fc   = (const float*)d_in[15];
  const float* thr    = (const float*)d_in[16];
  const float* leak   = (const float*)d_in[17];
  const float* mask2  = (const float*)d_in[18];
  const float* mask5  = (const float*)d_in[19];
  const float* mask9  = (const float*)d_in[20];
  const float* mask11 = (const float*)d_in[21];
  const float* mask13 = (const float*)d_in[22];
  const float* mask15 = (const float*)d_in[23];

  float* out = (float*)d_out;            // [32,10] accumulator

  // ---- workspace carve ----
  const int memTotal = 3 * 2097152 + 2 * 524288 + 2 * 262144 + 2 * 131072 + 2 * 65536;
  char* p = (char*)d_ws;
  float*  D  = (float*)p;  p += (size_t)2097152 * 4;
  __bf16* A  = (__bf16*)p; p += (size_t)2097152 * 2;
  __bf16* Bb = (__bf16*)p; p += (size_t)2097152 * 2;
  __bf16* xb = (__bf16*)p; p += (size_t)98304 * 2;     // bf16 copy of x
  float*  m0 = (float*)p;  p += (size_t)memTotal * 4;
  float* m1  = m0 + 2097152;
  float* m2  = m1 + 2097152;
  float* m3  = m2 + 2097152;   // B*64*16*16 = 524288
  float* m4  = m3 + 524288;
  float* m5  = m4 + 524288;    // B*128*8*8 = 262144
  float* m6  = m5 + 262144;
  float* m7  = m6 + 262144;    // B*256*4*4 = 131072
  float* m8  = m7 + 131072;
  float* m9  = m8 + 131072;    // B*512*2*2 = 65536
  float* m10 = m9 + 65536;
  __bf16* wb = (__bf16*)p;     // bf16 K-padded weights, carved below

  // ---- once-per-launch conversions (deterministic, capture-safe) ----
  auto cvtw = [&](int which, int Cout_, int K_) -> __bf16* {
    int Kpad = (K_ + 63) & ~63;
    int n = Cout_ * Kpad;
    __bf16* dst = wb;
    cvt_w_kernel<<<(n + 255) / 256, 256, 0, stream>>>(
        (const float*)d_in[which], dst, K_, Kpad, n);
    wb += n;
    return dst;
  };
  __bf16* wb_pre0 = cvtw(1,  64,  27);
  __bf16* wb_pre1 = cvtw(2,  64,  576);
  __bf16* wb_pre2 = cvtw(3,  64,  576);
  __bf16* wb1a    = cvtw(4,  64,  576);
  __bf16* wb1b    = cvtw(5,  64,  576);
  __bf16* wb2a    = cvtw(6,  128, 576);
  __bf16* wb2b    = cvtw(7,  128, 1152);
  __bf16* wb2i    = cvtw(8,  128, 64);
  __bf16* wb3a    = cvtw(9,  256, 1152);
  __bf16* wb3b    = cvtw(10, 256, 2304);
  __bf16* wb3i    = cvtw(11, 256, 128);
  __bf16* wb4a    = cvtw(12, 512, 2304);
  __bf16* wb4b    = cvtw(13, 512, 4608);
  __bf16* wb4i    = cvtw(14, 512, 256);
  // x -> bf16 (no padding: K == Kpad == total)
  cvt_w_kernel<<<(98304 + 255) / 256, 256, 0, stream>>>(x, xb, 98304, 98304, 98304);

  // Deterministic init: zero all state we integrate into.
  zero_kernel<<<(memTotal + 255) / 256, 256, 0, stream>>>(m0, memTotal);
  zero_kernel<<<2, 256, 0, stream>>>(out, BATCH * 10);

  // conv(in, w, out, Cin, log2(Hin), Cout, log2(Hout), KH, stride, pad, acc)
  auto conv = [&](const __bf16* in, const __bf16* w, float* o,
                  int Cin, int wish, int Cout, int wsh,
                  int KH, int s, int pd, int acc) {
    dim3 grid((BATCH << (2 * wsh)) / 64, Cout / 32);
    if (KH == 3)
      conv_wmma_kernel<3><<<grid, 256, 0, stream>>>(in, w, o, Cin, wish, Cout, wsh, s, pd, acc);
    else
      conv_wmma_kernel<1><<<grid, 256, 0, stream>>>(in, w, o, Cin, wish, Cout, wsh, s, pd, acc);
  };
  auto lif = [&](float* m, const float* d, const __bf16* r, const float* mk,
                 __bf16* o, int layer, int n) {
    lif_kernel<<<(n + 255) / 256, 256, 0, stream>>>(m, d, r, mk, o, thr, leak, layer, n);
  };

  for (int t = 0; t < TSTEPS; ++t) {
    // --- pre_process (32x32 spatial) ---
    conv(xb, wb_pre0, D, 3,  5, 64, 5, 3, 1, 1, 0);
    lif(m0, D, nullptr, mask2, A, 0, 2097152);
    conv(A,  wb_pre1, D, 64, 5, 64, 5, 3, 1, 1, 0);
    lif(m1, D, nullptr, mask5, Bb, 1, 2097152);
    conv(Bb, wb_pre2, D, 64, 5, 64, 5, 3, 1, 1, 0);
    lif(m2, D, nullptr, nullptr, A, 2, 2097152);
    avgpool2_kernel<<<(524288 + 255) / 256, 256, 0, stream>>>(A, Bb, 64, 32, 32, 524288);
    // --- layer1 (identity shortcut, 16x16), inp = Bb ---
    conv(Bb, wb1a, D, 64, 4, 64, 4, 3, 1, 1, 0);
    lif(m3, D, nullptr, mask9, A, 3, 524288);
    conv(A,  wb1b, D, 64, 4, 64, 4, 3, 1, 1, 0);
    lif(m4, D, /*res=*/Bb, nullptr, A, 4, 524288);
    // --- layer2 (stride-2, 1x1 conv shortcut, ->8x8), inp = A ---
    conv(A,  wb2a, D, 64, 4, 128, 3, 3, 2, 1, 0);
    lif(m5, D, nullptr, mask11, Bb, 5, 262144);
    conv(Bb, wb2b, D, 128, 3, 128, 3, 3, 1, 1, 0);
    conv(A,  wb2i, D, 64, 4, 128, 3, 1, 2, 0, 1);
    lif(m6, D, nullptr, nullptr, A, 6, 262144);
    // --- layer3 (->4x4), inp = A ---
    conv(A,  wb3a, D, 128, 3, 256, 2, 3, 2, 1, 0);
    lif(m7, D, nullptr, mask13, Bb, 7, 131072);
    conv(Bb, wb3b, D, 256, 2, 256, 2, 3, 1, 1, 0);
    conv(A,  wb3i, D, 128, 3, 256, 2, 1, 2, 0, 1);
    lif(m8, D, nullptr, nullptr, A, 8, 131072);
    // --- layer4 (->2x2), inp = A ---
    conv(A,  wb4a, D, 256, 2, 512, 1, 3, 2, 1, 0);
    lif(m9, D, nullptr, mask15, Bb, 9, 65536);
    conv(Bb, wb4b, D, 512, 1, 512, 1, 3, 1, 1, 0);
    conv(A,  wb4i, D, 256, 2, 512, 1, 1, 2, 0, 1);
    lif(m10, D, nullptr, nullptr, A, 10, 65536);
    // --- classifier current accumulation ---
    fc_accum_kernel<<<(BATCH * 10 + 63) / 64, 64, 0, stream>>>(A, w_fc, out);
  }
}
